// GCNLayer_27951647162470
// MI455X (gfx1250) — compile-verified, matrix operand order checked by reading, not run on hardware
//
#include <hip/hip_runtime.h>
#include <hip/hip_bf16.h>

// ---------------------------------------------------------------------------
// GCN layer, MI455X (gfx1250, wave32, WMMA).
//   out = relu( (D^-1/2 (A+I) D^-1/2) @ X @ W^T + b )
// adj is 8192x8192 fp32 (256 MiB) -> memory bound: two streaming passes over adj
// ~= 23us at 23.3 TB/s. Aggregation GEMM (17.2 GFLOP) uses
// v_wmma_f32_16x16x32_bf16 (adj is exactly 0/1 in bf16; scaled features are
// hi/lo bf16 split for ~16-bit effective mantissa). adj loads are non-temporal
// so the 256 MiB stream does not evict the L2-resident 4 MB gT reuse set.
// ---------------------------------------------------------------------------

#define NN      8192
#define IN_DIM  128
#define OUT_DIM 128

typedef __attribute__((ext_vector_type(16))) __bf16 v16bf;
typedef __attribute__((ext_vector_type(8)))  __bf16 v8bf;
typedef __attribute__((ext_vector_type(8)))  float  v8f;
typedef __attribute__((ext_vector_type(4)))  float  f4;

// ---------------------------------------------------------------------------
// Kernel 1: one wave per row; row_sum(adj) + diagonal 1 -> d = rsqrt(rs + 1).
// rs >= 1 so the reference's inf-guard is vacuous. NT loads: pure stream.
// ---------------------------------------------------------------------------
__global__ void __launch_bounds__(256)
rowsum_d_kernel(const float* __restrict__ adj, float* __restrict__ dvec) {
    const int wave = (int)((blockIdx.x * blockDim.x + threadIdx.x) >> 5);
    const int lane = (int)(threadIdx.x & 31);
    if (wave >= NN) return;
    const float* row = adj + (size_t)wave * NN;
    float s = 0.0f;
    for (int k = lane * 4; k < NN; k += 32 * 4) {
        f4 v = __builtin_nontemporal_load((const f4*)(row + k));
        s += v[0] + v[1] + v[2] + v[3];
    }
    #pragma unroll
    for (int off = 16; off > 0; off >>= 1)
        s += __shfl_xor(s, off, 32);
    if (lane == 0) dvec[wave] = rsqrtf(s + 1.0f);
}

// ---------------------------------------------------------------------------
// Kernel 2: g = d[j] * X[j][c], stored TRANSPOSED as gT[c][j] in bf16 hi/lo
// so the GEMM's B-fragments are contiguous 16B loads.
// ---------------------------------------------------------------------------
__global__ void __launch_bounds__(256)
build_gT_kernel(const float* __restrict__ feat, const float* __restrict__ dvec,
                __bf16* __restrict__ gT_hi, __bf16* __restrict__ gT_lo) {
    const int tid = (int)(blockIdx.x * blockDim.x + threadIdx.x); // = c*NN + j
    const int c = tid >> 13;          // / 8192
    const int j = tid & (NN - 1);
    if (c >= IN_DIM) return;
    const float g  = dvec[j] * feat[(size_t)j * IN_DIM + c];
    const __bf16 hi = (__bf16)g;
    const __bf16 lo = (__bf16)(g - (float)hi);
    gT_hi[tid] = hi;
    gT_lo[tid] = lo;
}

// ---------------------------------------------------------------------------
// Kernel 3: h[i][c] = d_i * ( sum_j adj[i][j]*g[j][c] + d_i*X[i][c] )
// Block = 8 waves = one 16-row M-tile covering all 128 columns (adj read once).
// Wave w -> N-tile [16w,16w+16). Per K=32 step: A = adj tile (fp32->bf16,
// exact since adj in {0,1}); B = gT_hi / gT_lo; two WMMAs into one f32 acc.
//
// Fragment layouts (ISA 7.12.2, wave32):
//   A 16x32 bf16: lane&15 = M row, lane>>4 = K-half; vec elems 0..7 = K
//     half*8..+7, elems 8..15 = K 16+half*8..+7.
//   B 32x16 bf16: lane&15 = N col, same K grouping (gT makes these contiguous).
//   C/D 16x16 f32: lane&15 = N col, acc[r] = row r + 8*(lane>>4).
// ---------------------------------------------------------------------------
__global__ void __launch_bounds__(256)
aggregate_wmma_kernel(const float*  __restrict__ adj,
                      const float*  __restrict__ feat,
                      const float*  __restrict__ dvec,
                      const __bf16* __restrict__ gT_hi,
                      const __bf16* __restrict__ gT_lo,
                      float* __restrict__ h) {
    const int wv   = (int)(threadIdx.x >> 5);   // 0..7 : N-tile
    const int lane = (int)(threadIdx.x & 31);
    const int half = lane >> 4;
    const int l15  = lane & 15;
    const int row0 = (int)blockIdx.x * 16;
    const int col  = wv * 16 + l15;

    const float*  arow = adj   + (size_t)(row0 + l15) * NN + half * 8;
    const __bf16* bhi  = gT_hi + (size_t)col * NN + half * 8;
    const __bf16* blo  = gT_lo + (size_t)col * NN + half * 8;

    v8f acc = {};

    #pragma unroll 2
    for (int k0 = 0; k0 < NN; k0 += 32) {
        // ---- issue ALL loads first (one clause, one wait) ----
        // A: adjacency, streamed, non-temporal (keep gT resident in L2)
        f4 a0 = __builtin_nontemporal_load((const f4*)(arow + k0));
        f4 a1 = __builtin_nontemporal_load((const f4*)(arow + k0 + 4));
        f4 a2 = __builtin_nontemporal_load((const f4*)(arow + k0 + 16));
        f4 a3 = __builtin_nontemporal_load((const f4*)(arow + k0 + 20));
        // B: transposed bf16 hi/lo (L2-resident, reused 512x)
        v8bf bh0 = *(const v8bf*)(bhi + k0);
        v8bf bh1 = *(const v8bf*)(bhi + k0 + 16);
        v8bf bl0 = *(const v8bf*)(blo + k0);
        v8bf bl1 = *(const v8bf*)(blo + k0 + 16);

        // speculative stream-ahead prefetch (SYS scope: dropped if unmapped)
        __builtin_prefetch(arow + k0 + 1024, 0, 0);

        // ---- pack fragments ----
        v16bf a;
        #pragma unroll
        for (int e = 0; e < 4; ++e) {
            a[e]      = (__bf16)a0[e];
            a[4 + e]  = (__bf16)a1[e];
            a[8 + e]  = (__bf16)a2[e];
            a[12 + e] = (__bf16)a3[e];
        }
        v16bf bh, bl;
        #pragma unroll
        for (int e = 0; e < 8; ++e) {
            bh[e] = bh0[e]; bh[8 + e] = bh1[e];
            bl[e] = bl0[e]; bl[8 + e] = bl1[e];
        }

        acc = __builtin_amdgcn_wmma_f32_16x16x32_bf16(
                  false, a, false, bh, (short)0, acc, false, false);
        acc = __builtin_amdgcn_wmma_f32_16x16x32_bf16(
                  false, a, false, bl, (short)0, acc, false, false);
    }

    // Epilogue: diagonal of (A+I) contributes g_i = d_i*x_i, then scale by d_i.
    #pragma unroll
    for (int r = 0; r < 8; ++r) {
        const int i   = row0 + r + half * 8;
        const float di   = dvec[i];
        const float diag = di * feat[(size_t)i * IN_DIM + col];
        h[(size_t)i * IN_DIM + col] = di * (acc[r] + diag);
    }
}

// ---------------------------------------------------------------------------
// Kernel 4: out = relu(h @ W^T + b). K = 128 (4 WMMA steps). h and W are fp32,
// converted on the fly with hi/lo bf16 split; 3 WMMAs per step (drop lo*lo).
// B column o needs W[o][k] contiguous in k == W's native row-major layout.
// ---------------------------------------------------------------------------
__global__ void __launch_bounds__(256)
linear_relu_wmma_kernel(const float* __restrict__ h,
                        const float* __restrict__ W,
                        const float* __restrict__ bias,
                        float* __restrict__ out) {
    const int wv   = (int)(threadIdx.x >> 5);
    const int lane = (int)(threadIdx.x & 31);
    const int half = lane >> 4;
    const int l15  = lane & 15;
    const int row0 = (int)blockIdx.x * 16;
    const int col  = wv * 16 + l15;               // output feature o

    const float* hrow = h + (size_t)(row0 + l15) * IN_DIM + half * 8;
    const float* wrow = W + (size_t)col * IN_DIM + half * 8;

    v8f acc = {};

    #pragma unroll
    for (int k0 = 0; k0 < IN_DIM; k0 += 32) {
        f4 ha0 = *(const f4*)(hrow + k0);
        f4 ha1 = *(const f4*)(hrow + k0 + 4);
        f4 ha2 = *(const f4*)(hrow + k0 + 16);
        f4 ha3 = *(const f4*)(hrow + k0 + 20);
        f4 wb0 = *(const f4*)(wrow + k0);
        f4 wb1 = *(const f4*)(wrow + k0 + 4);
        f4 wb2 = *(const f4*)(wrow + k0 + 16);
        f4 wb3 = *(const f4*)(wrow + k0 + 20);

        v16bf ahi, alo, bhi, blo;
        #pragma unroll
        for (int e = 0; e < 4; ++e) {
            float av[4] = { ha0[e], ha1[e], ha2[e], ha3[e] };
            float bv[4] = { wb0[e], wb1[e], wb2[e], wb3[e] };
            #pragma unroll
            for (int gidx = 0; gidx < 4; ++gidx) {
                const int slot = gidx * 4 + e;
                __bf16 ah = (__bf16)av[gidx];
                ahi[slot] = ah;
                alo[slot] = (__bf16)(av[gidx] - (float)ah);
                __bf16 bh = (__bf16)bv[gidx];
                bhi[slot] = bh;
                blo[slot] = (__bf16)(bv[gidx] - (float)bh);
            }
        }

        acc = __builtin_amdgcn_wmma_f32_16x16x32_bf16(
                  false, ahi, false, bhi, (short)0, acc, false, false);
        acc = __builtin_amdgcn_wmma_f32_16x16x32_bf16(
                  false, ahi, false, blo, (short)0, acc, false, false);
        acc = __builtin_amdgcn_wmma_f32_16x16x32_bf16(
                  false, alo, false, bhi, (short)0, acc, false, false);
    }

    const float bo = bias[col];
    #pragma unroll
    for (int r = 0; r < 8; ++r) {
        const int i = row0 + r + half * 8;
        float v = acc[r] + bo;
        out[(size_t)i * OUT_DIM + col] = v > 0.0f ? v : 0.0f;
    }
}

// ---------------------------------------------------------------------------
// Launch. Workspace layout (bytes):
//   [0, 32K)          d        : 8192 f32
//   [32K, +2M)        gT_hi    : 128*8192 bf16 (transposed)
//   [.. +2M)          gT_lo    : 128*8192 bf16
//   [.. +4M)          h        : 8192*128 f32
// Total ~8.03 MB.
// ---------------------------------------------------------------------------
extern "C" void kernel_launch(void* const* d_in, const int* in_sizes, int n_in,
                              void* d_out, int out_size, void* d_ws, size_t ws_size,
                              hipStream_t stream) {
    (void)in_sizes; (void)n_in; (void)out_size; (void)ws_size;

    const float* adj  = (const float*)d_in[0];
    const float* feat = (const float*)d_in[1];
    const float* W    = (const float*)d_in[2];
    const float* bias = (const float*)d_in[3];
    float* out = (float*)d_out;

    char* ws = (char*)d_ws;
    float*  dvec  = (float*)ws;
    __bf16* gT_hi = (__bf16*)(ws + (32u << 10));
    __bf16* gT_lo = (__bf16*)(ws + (32u << 10) + (2u << 20));
    float*  h     = (float*)(ws + (32u << 10) + (4u << 20));

    // 1) degrees -> d = rsqrt(rowsum + 1)      (reads adj once: 256 MiB, NT)
    rowsum_d_kernel<<<NN / 8, 256, 0, stream>>>(adj, dvec);
    // 2) transposed bf16 hi/lo scaled features (tiny)
    build_gT_kernel<<<(NN * IN_DIM) / 256, 256, 0, stream>>>(feat, dvec, gT_hi, gT_lo);
    // 3) aggregation GEMM via WMMA            (reads adj once more: 256 MiB, NT)
    aggregate_wmma_kernel<<<NN / 16, 256, 0, stream>>>(adj, feat, dvec, gT_hi, gT_lo, h);
    // 4) linear + bias + relu via WMMA        (tiny)
    linear_relu_wmma_kernel<<<NN / 16, 256, 0, stream>>>(h, W, bias, out);
}
